// FixedRC_56341380989595
// MI455X (gfx1250) — matrix operand order repaired and checked by reading, not compiled, not run on hardware
//
#include <hip/hip_runtime.h>

typedef __bf16 bf16_t;
typedef __attribute__((ext_vector_type(16))) __bf16 v16bf;
typedef __attribute__((ext_vector_type(8)))  float  v8f;
typedef __attribute__((ext_vector_type(4)))  float  v4f;
typedef __attribute__((ext_vector_type(4)))  unsigned int u32x4;
typedef __attribute__((ext_vector_type(8)))  int          i32x8;
typedef __attribute__((ext_vector_type(4)))  int          i32x4;

#define N_NODES 16384
#define DFEAT   64
#define KSTAGE  128                       // K elements per TDM stage
#define NSTAGES (N_NODES / KSTAGE)        // 128 stages
#define BPAD    8                         // 8 bf16 = 16B pad per 256B row
#define BSTRIDE (KSTAGE + BPAD)           // 136 bf16 = 272B LDS column stride

// ---------------------------------------------------------------------------
// Phase 1: YbT[d][row] = bf16( (X @ W_agg)[row][d] )   (transposed for WMMA-B)
// ---------------------------------------------------------------------------
__global__ __launch_bounds__(256)
void gcn_phase1_xw(const float* __restrict__ X, const float* __restrict__ W,
                   bf16_t* __restrict__ YbT) {
  __shared__ float sW[DFEAT * DFEAT];   // 16 KB
  __shared__ float sX[16 * DFEAT];      // 4 KB
  const int tid = threadIdx.x;
  const int rowBase = blockIdx.x * 16;

  for (int i = tid; i < DFEAT * DFEAT; i += 256) sW[i] = W[i];
  for (int i = tid; i < 16 * DFEAT;    i += 256) sX[i] = X[(size_t)rowBase * DFEAT + i];
  __syncthreads();

  for (int e = tid; e < 16 * DFEAT; e += 256) {
    const int r = e >> 6;
    const int d = e & 63;
    float acc = 0.f;
#pragma unroll
    for (int k = 0; k < DFEAT; ++k)
      acc = __builtin_fmaf(sX[r * DFEAT + k], sW[k * DFEAT + d], acc);
    YbT[(size_t)d * N_NODES + rowBase + r] = (bf16_t)acc;
  }
}

// ---------------------------------------------------------------------------
// TDM: copy a [DFEAT rows x KSTAGE cols] bf16 tile of YbT (row = feature
// column, contiguous along K) into LDS, padding 16B per 256B row so the LDS
// stride is BSTRIDE bf16. D# per CDNA5 ISA ch.8.
// ---------------------------------------------------------------------------
__device__ __forceinline__ void tdm_load_B(const bf16_t* gsrc, unsigned lds_byte_addr) {
  const unsigned long long ga = (unsigned long long)(size_t)gsrc;
  u32x4 g0;
  g0[0] = 1u;                                        // count=1 (valid), no gather
  g0[1] = lds_byte_addr;                             // lds_addr
  g0[2] = (unsigned)(ga & 0xFFFFFFFFu);              // global_addr[31:0]
  g0[3] = (unsigned)((ga >> 32) & 0x01FFFFFFu)       // global_addr[56:32]
        | (2u << 30);                                // type = 2 ("image")
  i32x8 g1;
  g1[0] = (1 << 16)                                  // data_size = 2 bytes
        | (1 << 20)                                  // pad_enable
        | (5 << 22)                                  // pad_interval: 64 dwords (256B)
        | (3 << 25);                                 // pad_amount: 4 dwords (16B)
  g1[1] = (int)(N_NODES << 16);                      // tensor_dim0[15:0] in [31:16]
  g1[2] = (int)(DFEAT << 16);                        // tensor_dim0 hi=0 | tensor_dim1[15:0]
  g1[3] = (int)(KSTAGE << 16);                       // tensor_dim1 hi=0 | tile_dim0
  g1[4] = DFEAT;                                     // tile_dim1 | tile_dim2=0
  g1[5] = N_NODES;                                   // tensor_dim0_stride[31:0]
  g1[6] = 0;                                         // stride hi | dim1_stride lo (unused, 2D)
  g1[7] = 0;
  const i32x4 gz4 = {0, 0, 0, 0};                    // groups 2/3 unused (2D tile)
  const i32x8 gz8 = {0, 0, 0, 0, 0, 0, 0, 0};        // trailing group (unused)
  __builtin_amdgcn_tensor_load_to_lds(g0, g1, gz4, gz4, gz8, 0);
}

// ---------------------------------------------------------------------------
// Phase 2: out = w*relu(A_hat @ Y) + (1-w)*X
// 4 waves / 128-thread block, one 16-row tile per wave. B double-buffered in
// LDS via TDM (tensor_load_to_lds + s_wait_tensorcnt), A streamed fp32 from
// HBM and converted to bf16 in registers, v_wmma_f32_16x16x32_bf16 core.
// ---------------------------------------------------------------------------
__global__ __launch_bounds__(128)
void gcn_phase2_agg(const float* __restrict__ A, const bf16_t* __restrict__ YbT,
                    const float* __restrict__ w, const float* __restrict__ X,
                    float* __restrict__ out) {
  __shared__ __align__(32) bf16_t sB[2][DFEAT * BSTRIDE];   // 2 x 17408 B

  const int lane = threadIdx.x & 31;
  const int wave = threadIdx.x >> 5;
  const int tile = blockIdx.x * 4 + wave;    // 0..1023
  const int m0   = tile * 16;
  const int n    = lane & 15;
  const int half = lane >> 4;

  // A fragment rows (16-bit A 16x32, contiguous k order shared with B): lane holds row m0+n,
  // contiguous K run of 16 at half*16.
  const float* Arow = A + (size_t)(m0 + n) * N_NODES + half * 16;

  // Escape the LDS base with a memory clobber: TDM writes LDS behind the
  // compiler's back, so forbid "never stored -> undef" folding of sB reads.
  asm volatile("" :: "s"((unsigned)(size_t)&sB[0][0]) : "memory");

  const bool issuer = (wave == 0);
  if (issuer) tdm_load_B(YbT, (unsigned)(size_t)&sB[0][0]);
  __builtin_amdgcn_s_wait_tensorcnt(0);
  __syncthreads();

  v8f acc0 = {}, acc1 = {}, acc2 = {}, acc3 = {};

  for (int s = 0; s < NSTAGES; ++s) {
    // Kick the DMA for the next stage into the other buffer, then compute.
    if (issuer && (s + 1) < NSTAGES)
      tdm_load_B(YbT + (size_t)(s + 1) * KSTAGE,
                 (unsigned)(size_t)&sB[(s + 1) & 1][0]);

    const bf16_t* sb  = &sB[s & 1][0];
    const float*  apb = Arow + s * KSTAGE;

#pragma unroll
    for (int kk = 0; kk < KSTAGE; kk += 32) {
      const float* ap = apb + kk;
      __builtin_prefetch(ap + 1024, 0, 0);           // stream A ahead into GL2

      v4f f0 = *(const v4f*)(ap + 0);
      v4f f1 = *(const v4f*)(ap + 4);
      v4f f2 = *(const v4f*)(ap + 8);
      v4f f3 = *(const v4f*)(ap + 12);

      v16bf a;
#pragma unroll
      for (int i = 0; i < 4; ++i) {
        a[i]      = (bf16_t)f0[i];
        a[4 + i]  = (bf16_t)f1[i];
        a[8 + i]  = (bf16_t)f2[i];
        a[12 + i] = (bf16_t)f3[i];
      }

      // B fragments from LDS (16-bit B 32x16 layout): column n of each
      // 16-wide N tile, contiguous 32B K-run at kk + half*16.
      const bf16_t* bbase = sb + kk + half * 16;
      v16bf b0 = *(const v16bf*)(bbase + (size_t)( 0 + n) * BSTRIDE);
      v16bf b1 = *(const v16bf*)(bbase + (size_t)(16 + n) * BSTRIDE);
      v16bf b2 = *(const v16bf*)(bbase + (size_t)(32 + n) * BSTRIDE);
      v16bf b3 = *(const v16bf*)(bbase + (size_t)(48 + n) * BSTRIDE);

      acc0 = __builtin_amdgcn_wmma_f32_16x16x32_bf16(false, a, false, b0, (short)0, acc0, false, false);
      acc1 = __builtin_amdgcn_wmma_f32_16x16x32_bf16(false, a, false, b1, (short)0, acc1, false, false);
      acc2 = __builtin_amdgcn_wmma_f32_16x16x32_bf16(false, a, false, b2, (short)0, acc2, false, false);
      acc3 = __builtin_amdgcn_wmma_f32_16x16x32_bf16(false, a, false, b3, (short)0, acc3, false, false);
    }

    __builtin_amdgcn_s_wait_tensorcnt(0);  // no-op for non-issuing waves
    __syncthreads();                       // next-stage buffer visible to all
  }

  // Fused epilogue: C/D layout -> VGPR r holds row m0 + 8*half + r, col n.
  v4f wlo = *(const v4f*)(w + m0 + 8 * half);
  v4f whi = *(const v4f*)(w + m0 + 8 * half + 4);

#pragma unroll
  for (int r = 0; r < 8; ++r) {
    const int   row = m0 + 8 * half + r;
    const float wr  = (r < 4) ? wlo[r] : whi[r - 4];
    const float cr  = 1.0f - wr;
    const size_t base = (size_t)row * DFEAT + n;

    float v0 = fmaxf(acc0[r], 0.f);
    float v1 = fmaxf(acc1[r], 0.f);
    float v2 = fmaxf(acc2[r], 0.f);
    float v3 = fmaxf(acc3[r], 0.f);

    out[base +  0] = wr * v0 + cr * X[base +  0];
    out[base + 16] = wr * v1 + cr * X[base + 16];
    out[base + 32] = wr * v2 + cr * X[base + 32];
    out[base + 48] = wr * v3 + cr * X[base + 48];
  }
}

// ---------------------------------------------------------------------------
extern "C" void kernel_launch(void* const* d_in, const int* in_sizes, int n_in,
                              void* d_out, int out_size, void* d_ws, size_t ws_size,
                              hipStream_t stream) {
  const float* X    = (const float*)d_in[0];   // [16384 x 64]
  const float* Ahat = (const float*)d_in[1];   // [16384 x 16384]
  const float* w    = (const float*)d_in[2];   // [16384]
  const float* Wagg = (const float*)d_in[3];   // [64 x 64]
  float*       out  = (float*)d_out;           // [16384 x 64]

  bf16_t* YbT = (bf16_t*)d_ws;                 // [64 x 16384] bf16 = 2 MB scratch

  gcn_phase1_xw<<<N_NODES / 16, 256, 0, stream>>>(X, Wagg, YbT);
  // 1024 row-tiles, 4 waves (tiles) per 128-thread block -> 256 blocks
  gcn_phase2_agg<<<(N_NODES / 16) / 4, 128, 0, stream>>>(Ahat, YbT, w, X, out);
}
